// GConvNetBigGraph_40475771798170
// MI455X (gfx1250) — matrix operand order repaired and checked by reading, not compiled, not run on hardware
//
#include <hip/hip_runtime.h>
#include <hip/hip_bf16.h>

typedef __attribute__((ext_vector_type(16))) _Float16 v16h;
typedef __attribute__((ext_vector_type(8)))  float    v8f;

#define N_NODES  500000
#define N_EDGES  8000000
#define N_GRAPHS 64
#define IN_F     30
#define H1       500
#define H2       20
#define OUTF     4

#define NEG_BIG  (-3.402823466e38f)
#define LDB      34            // 17 dwords -> bank-conflict-free on 64 banks

__device__ __forceinline__ float fast_silu(float v) {
    // v * sigmoid(v) using v_rcp_f32 instead of IEEE divide
    return v * __builtin_amdgcn_rcpf(1.0f + __expf(-v));
}
__device__ __forceinline__ float fast_sigmoid(float v) {
    return __builtin_amdgcn_rcpf(1.0f + __expf(-v));
}

// ---------------- workspace layout (floats) ----------------
// [0,        500000)   deg_out -> norm_out
// [500000,  1000000)   deg_in  -> norm_in
// [1000000, 16000000)  agg [500000 x 30]
// [16000000,16032000)  pooled [64 x 500] (init -FLT_MAX)

__global__ __launch_bounds__(256) void init_ws_kernel(float* ws) {
    long i = (long)blockIdx.x * blockDim.x + threadIdx.x;
    if (i < 16000000L)       ws[i] = 0.0f;
    else if (i < 16032000L)  ws[i] = NEG_BIG;
}

__global__ __launch_bounds__(256) void degree_kernel(const int* __restrict__ src,
                                                     const int* __restrict__ dst,
                                                     float* __restrict__ deg_out,
                                                     float* __restrict__ deg_in) {
    int e = blockIdx.x * blockDim.x + threadIdx.x;
    if (e < N_EDGES) {
        atomicAdd(&deg_out[src[e]], 1.0f);
        atomicAdd(&deg_in [dst[e]], 1.0f);
    }
}

__global__ __launch_bounds__(256) void norm_kernel(float* __restrict__ deg_out,
                                                   float* __restrict__ deg_in) {
    int n = blockIdx.x * blockDim.x + threadIdx.x;
    if (n < N_NODES) {
        float dO = deg_out[n], dI = deg_in[n];
        deg_out[n] = (dO > 0.0f) ? __frsqrt_rn(dO) : 0.0f;
        deg_in [n] = (dI > 0.0f) ? __frsqrt_rn(dI) : 0.0f;
    }
}

// one wave32 per edge; lanes 0..29 carry the 30 features (coalesced 120B row read)
__global__ __launch_bounds__(256) void scatter_kernel(const float* __restrict__ feat,
                                                      const int*   __restrict__ src,
                                                      const int*   __restrict__ dst,
                                                      const float* __restrict__ norm_out,
                                                      float*       __restrict__ agg) {
    int wid  = (blockIdx.x * 256 + threadIdx.x) >> 5;   // edge id
    int lane = threadIdx.x & 31;
    if (wid >= N_EDGES) return;
    int s = src[wid];
    int d = dst[wid];
    if (lane < IN_F) {
        float v = feat[(long)s * IN_F + lane] * norm_out[s];
        atomicAdd(&agg[(long)d * IN_F + lane], v);
    }
}

// Fused: h = silu((agg * norm_in) @ W1 + b1), pooled = segment_max(h, graph_id).
// One wave per 16-row tile of agg; 32 N-tiles (500 cols padded to 512).
// Never materializes h (saves ~2 GB of HBM traffic vs the reference graph).
__global__ __launch_bounds__(256) void gemm_silu_pool_kernel(
        const float* __restrict__ agg,   const float* __restrict__ norm_in,
        const float* __restrict__ W1,    const float* __restrict__ b1,
        const int*   __restrict__ gid,   float*       __restrict__ pooled) {
    __shared__ _Float16 Bt[512 * LDB];   // W1^T, f16, zero-padded, ~34 KB
    __shared__ float    b1s[512];        // bias staged -> no VMEM in inner loop

    int tid = threadIdx.x;
    // stage W1^T (Bt[n][k] = W1[k*H1+n]) as f16, zero padding K 30->32, N 500->512
    for (int idx = tid; idx < 512 * 32; idx += 256) {
        int n = idx >> 5, k = idx & 31;
        float w = (k < IN_F && n < H1) ? W1[k * H1 + n] : 0.0f;
        Bt[n * LDB + k] = (_Float16)w;
    }
    for (int n = tid; n < 512; n += 256) b1s[n] = (n < H1) ? b1[n] : 0.0f;
    __syncthreads();

    int wave  = tid >> 5;
    int lane  = tid & 31;
    int tileM = blockIdx.x * 8 + wave;
    if (tileM >= (N_NODES / 16)) return;          // 31250 tiles total

    int l  = lane & 15;      // also output column within tile
    int hi = lane >> 4;
    int mbase = tileM * 16;
    int m = mbase + l;       // this lane's A-row

    // A fragment: 16-bit A 16x32 layout, K = (i/8)*16 + hi*8 + ((i/2)&3)*2 + (i&1)
    float nrm = norm_in[m];
    v16h afrag;
#pragma unroll
    for (int i = 0; i < 16; ++i) {
        int k = ((i >> 3) << 4) + hi * 8 + (((i >> 1) & 3) << 1) + (i & 1);
        float v = (k < IN_F) ? agg[(long)m * IN_F + k] * nrm : 0.0f;
        afrag[i] = (_Float16)v;
    }

    int g0  = gid[mbase];
    int g15 = gid[mbase + 15];
    bool uniform = (g0 == g15);   // graph_id sorted -> tile usually one graph

    for (int nt = 0; nt < 32; ++nt) {
        // B fragment: lane = column N=l, K = (i/8)*16 + hi*8 + (i&7)  (matches A)
        const _Float16* bp = &Bt[(nt * 16 + l) * LDB];
        v16h bfrag;
#pragma unroll
        for (int i = 0; i < 16; ++i) bfrag[i] = bp[((i >> 3) << 4) + hi * 8 + (i & 7)];

        v8f c = {};
        c = __builtin_amdgcn_wmma_f32_16x16x32_f16(
                /*neg_a=*/false, afrag, /*neg_b=*/false, bfrag,
                /*c_mod=*/(short)0, c, /*reuse_a=*/false, /*reuse_b=*/false);

        int col = nt * 16 + l;
        if (col < H1) {                       // lanes l and l+16 agree -> safe shfl
            float bias = b1s[col];
            if (uniform) {
                float mx = NEG_BIG;
#pragma unroll
                for (int r = 0; r < 8; ++r) {
                    float v = fast_silu(c[r] + bias);
                    mx = fmaxf(mx, v);
                }
                mx = fmaxf(mx, __shfl_xor(mx, 16, 32)); // combine M-halves (same col)
                if (hi == 0) atomicMax(&pooled[g0 * H1 + col], mx);
            } else {
#pragma unroll
                for (int r = 0; r < 8; ++r) {
                    int row = mbase + r + 8 * hi;       // C/D: M = r + 8*hi
                    float v = fast_silu(c[r] + bias);
                    atomicMax(&pooled[gid[row] * H1 + col], v);
                }
            }
        }
    }
}

// tiny tail: z = silu(pooled @ W2 + b2); out = sigmoid(z @ W3 + b3)
__global__ __launch_bounds__(256) void head_kernel(const float* __restrict__ pooled,
                                                   const float* __restrict__ W2,
                                                   const float* __restrict__ b2,
                                                   const float* __restrict__ W3,
                                                   const float* __restrict__ b3,
                                                   float* __restrict__ out) {
    __shared__ float z[N_GRAPHS * H2];
    int tid = threadIdx.x;
    for (int idx = tid; idx < N_GRAPHS * H2; idx += 256) {
        int g = idx / H2, j = idx % H2;
        float s = b2[j];
        for (int f = 0; f < H1; ++f) s += pooled[g * H1 + f] * W2[f * H2 + j];
        z[idx] = fast_silu(s);
    }
    __syncthreads();
    int g = tid >> 2, o = tid & 3;          // 256 threads == 64*4 outputs
    float s = b3[o];
#pragma unroll
    for (int j = 0; j < H2; ++j) s += z[g * H2 + j] * W3[j * OUTF + o];
    out[tid] = fast_sigmoid(s);
}

extern "C" void kernel_launch(void* const* d_in, const int* in_sizes, int n_in,
                              void* d_out, int out_size, void* d_ws, size_t ws_size,
                              hipStream_t stream) {
    const float* features = (const float*)d_in[0];
    const int*   src      = (const int*)  d_in[1];
    const int*   dst      = (const int*)  d_in[2];
    const int*   gid      = (const int*)  d_in[3];
    // d_in[4] = num_graphs (scalar, == 64)
    const float* W1 = (const float*)d_in[5];
    const float* b1 = (const float*)d_in[6];
    const float* W2 = (const float*)d_in[7];
    const float* b2 = (const float*)d_in[8];
    const float* W3 = (const float*)d_in[9];
    const float* b3 = (const float*)d_in[10];
    float* out = (float*)d_out;

    float* ws       = (float*)d_ws;
    float* deg_out  = ws;                 // -> norm_out after norm_kernel
    float* deg_in   = ws + 500000L;       // -> norm_in
    float* agg      = ws + 1000000L;
    float* pooled   = ws + 16000000L;

    init_ws_kernel<<<(16032000 + 255) / 256, 256, 0, stream>>>(ws);
    degree_kernel<<<(N_EDGES + 255) / 256, 256, 0, stream>>>(src, dst, deg_out, deg_in);
    norm_kernel<<<(N_NODES + 255) / 256, 256, 0, stream>>>(deg_out, deg_in);
    scatter_kernel<<<N_EDGES / 8, 256, 0, stream>>>(features, src, dst, deg_out, agg);
    gemm_silu_pool_kernel<<<(N_NODES / 16 + 7) / 8, 256, 0, stream>>>(
        agg, deg_in, W1, b1, gid, pooled);
    head_kernel<<<1, 256, 0, stream>>>(pooled, W2, b2, W3, b3, out);
}
